// Correlation_Layer_TF_59347858096188
// MI455X (gfx1250) — matrix operand order repaired and checked by reading, not compiled, not run on hardware
//
#include <hip/hip_runtime.h>

typedef __attribute__((ext_vector_type(16))) _Float16 v16h;
typedef __attribute__((ext_vector_type(8)))  _Float16 v8h;
typedef __attribute__((ext_vector_type(8)))  float    v8f;
typedef __attribute__((ext_vector_type(4)))  float    v4f;

#define NTAPS  81   // 3*3*3*3 kernel taps
#define KPAIRS 41   // ceil(81/2); tap 81 zero-padded in wB

// Padded f16 input layout in d_ws:
//   x16[n][hp:34][wp:34][dp:34][fp:18][ci:16], halo rows are zeros.
// Tap (fh,fw,fd,ff) for output (h,w,d,f) reads (h+fh, w+fw, d+fd, f+ff):
// always in range -> zero predication in the hot loop.
#define XROWS    (2 * 34 * 34 * 34 * 18)          // rows of 16 ci
#define WB_HALF  (82 * 512)                       // f16 elements in weight image
#define WB_BYTES (WB_HALF * 2)                    // 83,968
#define X16_OFF  84992                            // 1KB-aligned, > WB_BYTES
// total d_ws requirement: 84,992 + XROWS*32 = 45,363,200 bytes (~43.3 MB)

// weights f32 [fh][fw][fd][ff][ci][co] -> f16 wB[tap][co][ci];
// tap 81 region is zeros (pad for the odd tap count).
__global__ void prep_weights(const float* __restrict__ k,
                             _Float16* __restrict__ wB) {
    int e = blockIdx.x * blockDim.x + threadIdx.x;   // over 82*512
    if (e >= 82 * 512) return;
    int t  = e >> 9;
    int co = (e >> 4) & 31;
    int ci = e & 15;
    _Float16 v = (_Float16)0.f;
    if (t < NTAPS) v = (_Float16)k[(t * 16 + ci) * 32 + co];
    wB[(t * 32 + co) * 16 + ci] = v;
}

// x f32 [n][h][w][d][f][ci] -> zero-padded f16 x16 (one thread per 16-ci row)
__global__ void prep_x(const float* __restrict__ x,
                       _Float16* __restrict__ x16) {
    int e = blockIdx.x * blockDim.x + threadIdx.x;
    if (e >= XROWS) return;
    int r  = e;
    const int fp = r % 18; r /= 18;
    const int dp = r % 34; r /= 34;
    const int wp = r % 34; r /= 34;
    const int hp = r % 34;
    const int n  = r / 34;
    v8h lo = {}, hi = {};
    const bool interior =
        (unsigned)(hp - 1) < 32u && (unsigned)(wp - 1) < 32u &&
        (unsigned)(dp - 1) < 32u && (unsigned)(fp - 1) < 16u;
    if (interior) {
        const float* p = x +
            (((((n * 32 + hp - 1) * 32 + wp - 1) * 32 + dp - 1) * 16 + fp - 1) * 16);
        const v4f q0 = ((const v4f*)p)[0];
        const v4f q1 = ((const v4f*)p)[1];
        const v4f q2 = ((const v4f*)p)[2];
        const v4f q3 = ((const v4f*)p)[3];
        lo[0]=(_Float16)q0.x; lo[1]=(_Float16)q0.y; lo[2]=(_Float16)q0.z; lo[3]=(_Float16)q0.w;
        lo[4]=(_Float16)q1.x; lo[5]=(_Float16)q1.y; lo[6]=(_Float16)q1.z; lo[7]=(_Float16)q1.w;
        hi[0]=(_Float16)q2.x; hi[1]=(_Float16)q2.y; hi[2]=(_Float16)q2.z; hi[3]=(_Float16)q2.w;
        hi[4]=(_Float16)q3.x; hi[5]=(_Float16)q3.y; hi[6]=(_Float16)q3.z; hi[7]=(_Float16)q3.w;
    }
    v8h* dst = (v8h*)(x16 + e * 16);
    dst[0] = lo;
    dst[1] = hi;
}

// Compile-time tap -> relative x16 element offset ((fh,fw,fd) voxel + ff row)
__device__ __forceinline__ constexpr int reltap(int t) {
    return ((t / 27) * 1156 + ((t / 9) % 3) * 34 + (t / 3) % 3) * 288
         + (t % 3) * 16;
}

// Block = 8 waves, one (n,h,w) d-line; each wave register-blocks 4 consecutive
// d voxels (full [f 0..15] x [co 0..31] tile each). Weights live in LDS.
// Per K-step: 4 ds_load_b128 (B, shared by 4 voxels) + 8 global_load_b128 (A)
// + 8 v_wmma_f32_16x16x32_f16. Fully unrolled; all offsets are immediates.
__global__ __launch_bounds__(256) void conv4d_wmma(
    const _Float16* __restrict__ x16, const _Float16* __restrict__ wB,
    const float* __restrict__ bias, float* __restrict__ out) {

    extern __shared__ _Float16 sB[];        // 82*512 f16 = 83,968 B

    // ---- cooperative weight-image copy into LDS (b128 granules)
    {
        const uint4* src = (const uint4*)wB;
        uint4*       dst = (uint4*)sB;
        for (int idx = threadIdx.x; idx < WB_BYTES / 16; idx += 256)
            dst[idx] = src[idx];
    }
    __syncthreads();

    const int lane = threadIdx.x & 31;
    const int wave = threadIdx.x >> 5;
    int v = blockIdx.x;                     // (n,h,w) : 0..2047
    const int w = v & 31; v >>= 5;
    const int h = v & 31; v >>= 5;
    const int n = v;
    const int dbase = wave * 4;             // this wave's 4 voxels

    const bool hiL = lane >= 16;
    const int  m   = lane & 15;             // A row = frame f
    const int  col = lane & 15;             // B/D column

    // A base: lane owns ci [0..7] (lo) / [8..15] (hi) of row m, voxel dbase.
    const _Float16* pA = x16
        + (((n * 34 + h) * 34 + w) * 34 + dbase) * 288
        + m * 16 + (hiL ? 8 : 0);
    // B in LDS: lane reads ci [0..7] (lo) / [8..15] (hi) of tap 2i, then of tap 2i+1.
    const int sOff = (hiL ? 8 : 0) + col * 16;   // element offset

    v8f c[4][2] = {};                       // [voxel][co-half]

    #pragma unroll
    for (int i = 0; i < KPAIRS; ++i) {
        const int t0 = 2 * i;
        const v16h b0 = __builtin_shufflevector(*(const v8h*)(sB + sOff + t0 * 512), *(const v8h*)(sB + sOff + t0 * 512 + 512), 0,1,2,3,4,5,6,7,8,9,10,11,12,13,14,15);        // co 0..15
        const v16h b1 = __builtin_shufflevector(*(const v8h*)(sB + sOff + t0 * 512 + 256), *(const v8h*)(sB + sOff + t0 * 512 + 768), 0,1,2,3,4,5,6,7,8,9,10,11,12,13,14,15);  // co 16..31
        #pragma unroll
        for (int vv = 0; vv < 4; ++vv) {
            const v8h f0 = *(const v8h*)(pA + vv * 288 + reltap(t0));
            v8h f1;
            if (t0 + 1 < NTAPS)
                f1 = *(const v8h*)(pA + vv * 288 + reltap(t0 + 1));
            else
                f1 = (v8h){};   // pad tap; its B rows are zero anyway
            const v16h a = __builtin_shufflevector(f0, f1,
                0,1,2,3,4,5,6,7,8,9,10,11,12,13,14,15);
            c[vv][0] = __builtin_amdgcn_wmma_f32_16x16x32_f16(
                false, a, false, b0, (short)0, c[vv][0], false, false);
            c[vv][1] = __builtin_amdgcn_wmma_f32_16x16x32_f16(
                false, a, false, b1, (short)0, c[vv][1], false, false);
        }
    }

    // ---- epilogue: bias + store per 16x16 f32 C/D layout
    // (lanes 0-15: N=lane, VGPR r -> M=r ; lanes 16-31: N=lane-16, M=8+r)
    const float bs0 = bias[col];
    const float bs1 = bias[col + 16];
    const int fbase = hiL ? 8 : 0;
    #pragma unroll
    for (int vv = 0; vv < 4; ++vv) {
        const int voxel = blockIdx.x * 32 + dbase + vv;
        const int obase = voxel * 512;      // [n,h,w,d] * F*Cout
        #pragma unroll
        for (int r = 0; r < 8; ++r) {
            out[obase + (fbase + r) * 32 + col]      = c[vv][0][r] + bs0;
            out[obase + (fbase + r) * 32 + col + 16] = c[vv][1][r] + bs1;
        }
    }
}

extern "C" void kernel_launch(void* const* d_in, const int* in_sizes, int n_in,
                              void* d_out, int out_size, void* d_ws, size_t ws_size,
                              hipStream_t stream) {
    (void)in_sizes; (void)n_in; (void)out_size; (void)ws_size;
    const float* x    = (const float*)d_in[0];
    const float* k    = (const float*)d_in[1];
    const float* bias = (const float*)d_in[2];
    float* out = (float*)d_out;

    _Float16* wB  = (_Float16*)d_ws;                       // 83,968 B
    _Float16* x16 = (_Float16*)((char*)d_ws + X16_OFF);    // ~43.2 MB padded f16 x

    prep_weights<<<(82 * 512 + 255) / 256, 256, 0, stream>>>(k, wB);
    prep_x<<<(XROWS + 255) / 256, 256, 0, stream>>>(x, x16);
    // one block per (n,h,w) d-line: 2*32*32 = 2048 blocks, 8 waves x 4 voxels
    conv4d_wmma<<<2048, 256, WB_BYTES, stream>>>(x16, wB, bias, out);
}
